// CausalSelfAttention_5437428597398
// MI455X (gfx1250) — compile-verified, matrix operand order checked by reading, not run on hardware
//
#include <hip/hip_runtime.h>
#include <hip/hip_bf16.h>
#include <math.h>

// Problem constants
#define BATCH 2
#define TSEQ  2048
#define CDIM  1024
#define NHEAD 16
#define HDIM  64
#define NTOK  (BATCH * TSEQ)   // 4096

typedef __attribute__((ext_vector_type(16))) __bf16 v16bf;
typedef __attribute__((ext_vector_type(8)))  float  v8f;

union ABFrag {
  unsigned short us[16];
  uint4 q[2];
  v16bf v;
};

static __device__ __forceinline__ unsigned short f32_to_bf16(float f) {
  unsigned int u = __builtin_bit_cast(unsigned int, f);
  u += 0x7FFFu + ((u >> 16) & 1u);   // round to nearest even
  return (unsigned short)(u >> 16);
}

// ---------------------------------------------------------------------------
// Stage 1/4: fp32 -> bf16 conversion
// ---------------------------------------------------------------------------
__global__ void cvt_f32_bf16(const float* __restrict__ in,
                             unsigned short* __restrict__ out, int n) {
  int i = blockIdx.x * blockDim.x + threadIdx.x;
  if (i < n) out[i] = f32_to_bf16(in[i]);
}

// ---------------------------------------------------------------------------
// Stage 2/5: GEMM  out = X @ W^T + b   (X:[4096,1024] bf16, W:[1024,1024] bf16)
// One wave computes a 16(M) x 64(N) strip: the A fragment is loaded once per
// k-step and reused across four 16x16 N-tiles -> 2.5 b128 loads per WMMA.
// MODE 0: bf16 row-major out          (Q, K projections)
// MODE 1: bf16 per-head transposed out Vt[((b*H+h)*HD+d)*T + t]  (V projection)
// MODE 2: fp32 row-major out          (final y projection)
// ---------------------------------------------------------------------------
template <int MODE>
__global__ __launch_bounds__(256)
void gemm16(const unsigned short* __restrict__ A,
            const unsigned short* __restrict__ W,
            const float* __restrict__ bias,
            void* __restrict__ outp) {
  const int wave  = (blockIdx.x * blockDim.x + threadIdx.x) >> 5;
  const int lane  = threadIdx.x & 31;
  const int mt    = wave >> 4;                 // 0..255  (M tiles)
  const int nq    = wave & 15;                 // 0..15   (64-wide N quads)
  const int n16   = lane & 15;
  const int half8 = (lane >> 4) << 3;          // 0 or 8

  // A fragment: row m = lane%16 ; K = i + (i>=8?8:0) + half8
  const unsigned short* aBase = A + (size_t)(mt * 16 + n16) * CDIM + half8;
  // B fragments (= W^T): column n of B = row n of W, contiguous over K
  const unsigned short* wBase = W + (size_t)(nq * 64 + n16) * CDIM + half8;

  v8f acc0 = {}, acc1 = {}, acc2 = {}, acc3 = {};
#pragma unroll 2
  for (int k0 = 0; k0 < CDIM; k0 += 32) {
    ABFrag a, b0, b1, b2, b3;
    a.q[0]  = *(const uint4*)(aBase + k0);
    a.q[1]  = *(const uint4*)(aBase + k0 + 16);
    b0.q[0] = *(const uint4*)(wBase + (size_t)0 * 16 * CDIM + k0);
    b0.q[1] = *(const uint4*)(wBase + (size_t)0 * 16 * CDIM + k0 + 16);
    b1.q[0] = *(const uint4*)(wBase + (size_t)1 * 16 * CDIM + k0);
    b1.q[1] = *(const uint4*)(wBase + (size_t)1 * 16 * CDIM + k0 + 16);
    b2.q[0] = *(const uint4*)(wBase + (size_t)2 * 16 * CDIM + k0);
    b2.q[1] = *(const uint4*)(wBase + (size_t)2 * 16 * CDIM + k0 + 16);
    b3.q[0] = *(const uint4*)(wBase + (size_t)3 * 16 * CDIM + k0);
    b3.q[1] = *(const uint4*)(wBase + (size_t)3 * 16 * CDIM + k0 + 16);
    acc0 = __builtin_amdgcn_wmma_f32_16x16x32_bf16(false, a.v, false, b0.v,
                                                   (short)0, acc0, false, false);
    acc1 = __builtin_amdgcn_wmma_f32_16x16x32_bf16(false, a.v, false, b1.v,
                                                   (short)0, acc1, false, false);
    acc2 = __builtin_amdgcn_wmma_f32_16x16x32_bf16(false, a.v, false, b2.v,
                                                   (short)0, acc2, false, false);
    acc3 = __builtin_amdgcn_wmma_f32_16x16x32_bf16(false, a.v, false, b3.v,
                                                   (short)0, acc3, false, false);
  }

  v8f accs[4] = {acc0, acc1, acc2, acc3};
#pragma unroll
  for (int t = 0; t < 4; ++t) {
    const int col = nq * 64 + t * 16 + n16;
    const float bval = bias[col];
    if (MODE == 0) {
      unsigned short* out = (unsigned short*)outp;
#pragma unroll
      for (int j = 0; j < 8; ++j) {
        int row = mt * 16 + j + half8;
        out[(size_t)row * CDIM + col] = f32_to_bf16(accs[t][j] + bval);
      }
    } else if (MODE == 1) {
      unsigned short* out = (unsigned short*)outp;
      const int hh = col >> 6;      // head (== nq here)
      const int d  = col & 63;      // head-dim
#pragma unroll
      for (int j = 0; j < 8; ++j) {
        int row = mt * 16 + j + half8;   // global token
        int bb  = row >> 11;
        int t2  = row & (TSEQ - 1);
        out[((size_t)(bb * NHEAD + hh) * HDIM + d) * TSEQ + t2] =
            f32_to_bf16(accs[t][j] + bval);
      }
    } else {
      float* out = (float*)outp;
#pragma unroll
      for (int j = 0; j < 8; ++j) {
        int row = mt * 16 + j + half8;
        out[(size_t)row * CDIM + col] = accs[t][j] + bval;
      }
    }
  }
}

// ---------------------------------------------------------------------------
// Stage 3: fused causal attention with online softmax.
// One wave handles a 16-query tile for one (b,h). Keys processed 32 at a time.
// Writes full att_scores (mask -> -inf) and accumulates O = softmax(S) @ V.
// ---------------------------------------------------------------------------
__global__ __launch_bounds__(128)
void attn_fused(const unsigned short* __restrict__ Qh,   // [4096,1024] bf16
                const unsigned short* __restrict__ Kh,   // [4096,1024] bf16
                const unsigned short* __restrict__ Vt,   // [B,H,HD,T]  bf16
                const float* __restrict__ bias,          // [H,T,T]
                float* __restrict__ attOut,              // [B,H,T,T]
                float* __restrict__ Obuf) {              // [4096,1024] f32
  __shared__ unsigned short pls[4][16 * 32];             // P staging per wave

  const int wslot = threadIdx.x >> 5;
  const int wave  = blockIdx.x * (blockDim.x >> 5) + wslot;
  const int lane  = threadIdx.x & 31;
  const int n16   = lane & 15;
  const int half8 = (lane >> 4) << 3;

  const int qt = wave & 127;          // query tile within sequence
  const int bh = wave >> 7;
  const int h  = bh & (NHEAD - 1);
  const int b  = bh >> 4;
  const int qt16 = qt * 16;

  // Q A-fragments for the two 32-wide K-steps of head dim 64
  const size_t qrowg = (size_t)(b * TSEQ + qt16 + n16) * CDIM + h * HDIM + half8;
  ABFrag aq[2];
#pragma unroll
  for (int s = 0; s < 2; ++s) {
    aq[s].q[0] = *(const uint4*)(Qh + qrowg + 32 * s);
    aq[s].q[1] = *(const uint4*)(Qh + qrowg + 32 * s + 16);
  }

  v8f o0 = {}, o1 = {}, o2 = {}, o3 = {};
  float mrow[8], lrow[8];
#pragma unroll
  for (int j = 0; j < 8; ++j) { mrow[j] = -3.0e38f; lrow[j] = 0.0f; }

  const float scale = 0.125f;   // 1/sqrt(64)
  const int    qmax     = qt16 + 15;
  const size_t biasBase = (size_t)h * TSEQ * TSEQ;
  const size_t attBase  = (size_t)(b * NHEAD + h) * TSEQ * TSEQ;
  const size_t vtBase   = (size_t)(b * NHEAD + h) * HDIM * TSEQ;
  unsigned short* myp = pls[wslot];

  for (int k0 = 0; k0 < TSEQ; k0 += 32) {
    if (k0 > qmax) {
      // entirely above the diagonal: att_scores = -inf, no compute
#pragma unroll
      for (int u = 0; u < 2; ++u) {
        const int key = k0 + u * 16 + n16;
#pragma unroll
        for (int j = 0; j < 8; ++j) {
          int qrow = qt16 + j + half8;
          attOut[attBase + (size_t)qrow * TSEQ + key] = -INFINITY;
        }
      }
      continue;
    }

    // ---- S = Q K^T for 32 keys (two 16-key subtiles) ----
    v8f sv[2];
#pragma unroll
    for (int u = 0; u < 2; ++u) {
      const int kb16 = k0 + u * 16;
      const size_t krowg =
          (size_t)(b * TSEQ + kb16 + n16) * CDIM + h * HDIM + half8;
      ABFrag bk0, bk1;
      bk0.q[0] = *(const uint4*)(Kh + krowg);
      bk0.q[1] = *(const uint4*)(Kh + krowg + 16);
      bk1.q[0] = *(const uint4*)(Kh + krowg + 32);
      bk1.q[1] = *(const uint4*)(Kh + krowg + 48);
      v8f s = {};
      s = __builtin_amdgcn_wmma_f32_16x16x32_bf16(false, aq[0].v, false, bk0.v,
                                                  (short)0, s, false, false);
      s = __builtin_amdgcn_wmma_f32_16x16x32_bf16(false, aq[1].v, false, bk1.v,
                                                  (short)0, s, false, false);
      const int key = kb16 + n16;
#pragma unroll
      for (int j = 0; j < 8; ++j) {
        int qrow = qt16 + j + half8;
        float sc = s[j] * scale + bias[biasBase + (size_t)qrow * TSEQ + key];
        bool ok  = (key <= qrow);
        attOut[attBase + (size_t)qrow * TSEQ + key] = ok ? sc : -INFINITY;
        sv[u][j] = ok ? sc : -3.0e38f;
      }
    }

    // ---- online softmax over these 32 keys ----
    float corr8[8];
#pragma unroll
    for (int j = 0; j < 8; ++j) {
      float tmax = fmaxf(sv[0][j], sv[1][j]);
#pragma unroll
      for (int msk = 1; msk < 16; msk <<= 1)
        tmax = fmaxf(tmax, __shfl_xor(tmax, msk, 32));
      float mnew = fmaxf(mrow[j], tmax);
      float p0 = __expf(sv[0][j] - mnew);
      float p1 = __expf(sv[1][j] - mnew);
      float rs = p0 + p1;
#pragma unroll
      for (int msk = 1; msk < 16; msk <<= 1)
        rs += __shfl_xor(rs, msk, 32);
      float corr = __expf(mrow[j] - mnew);
      lrow[j] = lrow[j] * corr + rs;
      mrow[j] = mnew;
      corr8[j] = corr;
      // stage P (bf16) in LDS in plain row-major [16 x 32]
      int prow = j + half8;
      myp[prow * 32 + n16]      = f32_to_bf16(p0);
      myp[prow * 32 + 16 + n16] = f32_to_bf16(p1);
    }
#pragma unroll
    for (int j = 0; j < 8; ++j) {
      o0[j] *= corr8[j]; o1[j] *= corr8[j];
      o2[j] *= corr8[j]; o3[j] *= corr8[j];
    }

    __builtin_amdgcn_wave_barrier();
    asm volatile("s_wait_dscnt 0" ::: "memory");

    // reload P as a 16x32 A-fragment (ds_load_b128 x2)
    ABFrag ap;
    ap.q[0] = *(const uint4*)(myp + n16 * 32 + half8);
    ap.q[1] = *(const uint4*)(myp + n16 * 32 + half8 + 16);

    // V B-fragments: Vt rows are contiguous over keys
    const unsigned short* vb =
        Vt + vtBase + (size_t)n16 * TSEQ + k0 + half8;
    ABFrag bv0, bv1, bv2, bv3;
    bv0.q[0] = *(const uint4*)(vb + (size_t)0  * 16 * TSEQ);
    bv0.q[1] = *(const uint4*)(vb + (size_t)0  * 16 * TSEQ + 16);
    bv1.q[0] = *(const uint4*)(vb + (size_t)1  * 16 * TSEQ);
    bv1.q[1] = *(const uint4*)(vb + (size_t)1  * 16 * TSEQ + 16);
    bv2.q[0] = *(const uint4*)(vb + (size_t)2  * 16 * TSEQ);
    bv2.q[1] = *(const uint4*)(vb + (size_t)2  * 16 * TSEQ + 16);
    bv3.q[0] = *(const uint4*)(vb + (size_t)3  * 16 * TSEQ);
    bv3.q[1] = *(const uint4*)(vb + (size_t)3  * 16 * TSEQ + 16);

    o0 = __builtin_amdgcn_wmma_f32_16x16x32_bf16(false, ap.v, false, bv0.v,
                                                 (short)0, o0, false, false);
    o1 = __builtin_amdgcn_wmma_f32_16x16x32_bf16(false, ap.v, false, bv1.v,
                                                 (short)0, o1, false, false);
    o2 = __builtin_amdgcn_wmma_f32_16x16x32_bf16(false, ap.v, false, bv2.v,
                                                 (short)0, o2, false, false);
    o3 = __builtin_amdgcn_wmma_f32_16x16x32_bf16(false, ap.v, false, bv3.v,
                                                 (short)0, o3, false, false);
    __builtin_amdgcn_wave_barrier();
  }

  // normalize and write O (fp32, merged-head layout [4096,1024])
#pragma unroll
  for (int j = 0; j < 8; ++j) {
    int row = qt16 + j + half8;
    float inv = 1.0f / lrow[j];
    size_t og = (size_t)(b * TSEQ + row) * CDIM + h * HDIM + n16;
    Obuf[og + 0]  = o0[j] * inv;
    Obuf[og + 16] = o1[j] * inv;
    Obuf[og + 32] = o2[j] * inv;
    Obuf[og + 48] = o3[j] * inv;
  }
}

// ---------------------------------------------------------------------------
// Launcher
// ---------------------------------------------------------------------------
extern "C" void kernel_launch(void* const* d_in, const int* in_sizes, int n_in,
                              void* d_out, int out_size, void* d_ws, size_t ws_size,
                              hipStream_t stream) {
  const float* q    = (const float*)d_in[0];
  const float* k    = (const float*)d_in[1];
  const float* v    = (const float*)d_in[2];
  const float* bias = (const float*)d_in[3];
  const float* Wq   = (const float*)d_in[4];
  const float* bq   = (const float*)d_in[5];
  const float* Wk   = (const float*)d_in[6];
  const float* bk   = (const float*)d_in[7];
  const float* Wv   = (const float*)d_in[8];
  const float* bv   = (const float*)d_in[9];
  const float* Wp   = (const float*)d_in[10];
  const float* bp   = (const float*)d_in[11];

  // workspace carve-up (256B aligned)
  size_t off = 0;
  auto take = [&](size_t bytes) -> void* {
    off = (off + 255) & ~(size_t)255;
    void* p = (char*)d_ws + off;
    off += bytes;
    return p;
  };
  const size_t XB = (size_t)NTOK * CDIM;        // 4M elements
  const size_t WB = (size_t)CDIM * CDIM;        // 1M elements
  unsigned short* qb  = (unsigned short*)take(XB * 2);
  unsigned short* kb  = (unsigned short*)take(XB * 2);
  unsigned short* vbuf= (unsigned short*)take(XB * 2);
  unsigned short* wqb = (unsigned short*)take(WB * 2);
  unsigned short* wkb = (unsigned short*)take(WB * 2);
  unsigned short* wvb = (unsigned short*)take(WB * 2);
  unsigned short* wpb = (unsigned short*)take(WB * 2);
  unsigned short* Qh  = (unsigned short*)take(XB * 2);
  unsigned short* Kh  = (unsigned short*)take(XB * 2);
  unsigned short* Vt  = (unsigned short*)take(XB * 2);
  float*          Obuf= (float*)take(XB * 4);
  unsigned short* Obf = (unsigned short*)take(XB * 2);

  float* yOut   = (float*)d_out;                           // [B,T,C]
  float* attOut = (float*)d_out + XB;                      // [B,H,T,T]

  // Stage 1: convert inputs + weights to bf16
  const int CT = 256;
  cvt_f32_bf16<<<(int)(XB / CT), CT, 0, stream>>>(q, qb, (int)XB);
  cvt_f32_bf16<<<(int)(XB / CT), CT, 0, stream>>>(k, kb, (int)XB);
  cvt_f32_bf16<<<(int)(XB / CT), CT, 0, stream>>>(v, vbuf, (int)XB);
  cvt_f32_bf16<<<(int)(WB / CT), CT, 0, stream>>>(Wq, wqb, (int)WB);
  cvt_f32_bf16<<<(int)(WB / CT), CT, 0, stream>>>(Wk, wkb, (int)WB);
  cvt_f32_bf16<<<(int)(WB / CT), CT, 0, stream>>>(Wv, wvb, (int)WB);
  cvt_f32_bf16<<<(int)(WB / CT), CT, 0, stream>>>(Wp, wpb, (int)WB);

  // Stage 2: projections (4096 waves of 16x64 strips -> 512 blocks of 256)
  const int GB = (NTOK / 16) * (CDIM / 64) / 8;
  gemm16<0><<<GB, 256, 0, stream>>>(qb,   wqb, bq, Qh);
  gemm16<0><<<GB, 256, 0, stream>>>(kb,   wkb, bk, Kh);
  gemm16<1><<<GB, 256, 0, stream>>>(vbuf, wvb, bv, Vt);

  // Stage 3: fused attention (4096 waves -> 1024 blocks of 128)
  attn_fused<<<BATCH * NHEAD * (TSEQ / 16) / 4, 128, 0, stream>>>(
      Qh, Kh, Vt, bias, attOut, Obuf);

  // Stage 4: O -> bf16
  cvt_f32_bf16<<<(int)(XB / CT), CT, 0, stream>>>(Obuf, Obf, (int)XB);

  // Stage 5: output projection straight into y (fp32)
  gemm16<2><<<GB, 256, 0, stream>>>(Obf, wpb, bp, yOut);
}
